// Attention_86131274154467
// MI455X (gfx1250) — matrix operand order repaired and checked by reading, not compiled, not run on hardware
//
#include <hip/hip_runtime.h>
#include <hip/hip_fp16.h>
#include <stdint.h>

typedef _Float16 f16;
typedef __attribute__((ext_vector_type(16))) _Float16 v16h;
typedef __attribute__((ext_vector_type(8)))  float    v8f;

union Frag16 { v16h h; uint32_t u[8]; uint2 u2[4]; uint4 u4[2]; };
union V4H8   { uint4 v; f16 h[8]; };

// ---- WMMA fragment layouts (CDNA5 ISA 7.12.2, wave32) ----
// A 16x32 f16: row = lane%16; lanes<16 hold K{0..7,16..23}, lanes>=16 K{8..15,24..31},
//              pairs contiguous -> chunks of 4 halves at k = hi*8 + {0,4,16,20}.
// B 32x16 f16: col = lane%16; same K order as A: chunks of 8 halves at k = hi*8 + {0,16}.
__device__ __forceinline__ void load_a_frag(const f16* rowp, int hi, Frag16& f) {
    const f16* p = rowp + hi * 8;
    f.u2[0] = *(const uint2*)(p + 0);
    f.u2[1] = *(const uint2*)(p + 4);
    f.u2[2] = *(const uint2*)(p + 16);
    f.u2[3] = *(const uint2*)(p + 20);
}
__device__ __forceinline__ void load_b_frag(const f16* rowp, int hi, Frag16& f) {
    const f16* p = rowp + hi * 8;
    f.u4[0] = *(const uint4*)(p + 0);
    f.u4[1] = *(const uint4*)(p + 16);
}

// ---- CDNA5 async global->LDS copy (ASYNCcnt path) ----
__device__ __forceinline__ void async_copy_b128(void* lds_ptr, const void* gptr) {
    // low 32 bits of a flat shared pointer == wave-relative LDS byte address
    uint32_t lds = (uint32_t)(uintptr_t)lds_ptr;
    asm volatile("global_load_async_to_lds_b128 %0, %1, off"
                 :: "v"(lds), "v"(gptr) : "memory");
}
__device__ __forceinline__ void wait_async0() {
    asm volatile("s_wait_asynccnt 0x0" ::: "memory");
}

// =====================================================================
// Elementwise f32 -> f16
// =====================================================================
__global__ __launch_bounds__(256) void cvt_f16_kernel(const float* __restrict__ in,
                                                      f16* __restrict__ out, int n) {
    int i = blockIdx.x * 256 + threadIdx.x;
    if (i < n) out[i] = (f16)in[i];
}

// =====================================================================
// Transpose + convert: in[K][N] f32 -> out[N][K] f16
// =====================================================================
__global__ __launch_bounds__(256) void transpose_cvt_kernel(const float* __restrict__ in,
                                                            f16* __restrict__ out,
                                                            int K, int N) {
    __shared__ f16 tile[32][33];
    int nb = blockIdx.x * 32, kb = blockIdx.y * 32;
    int tx = threadIdx.x & 31, ty = threadIdx.x >> 5;
#pragma unroll
    for (int i = 0; i < 32; i += 8)
        tile[ty + i][tx] = (f16)in[(size_t)(kb + ty + i) * N + (nb + tx)];
    __syncthreads();
#pragma unroll
    for (int i = 0; i < 32; i += 8)
        out[(size_t)(nb + ty + i) * K + (kb + tx)] = tile[tx][ty + i];
}

// =====================================================================
// WMMA GEMM: C[M,N] = A[M,K] * Bt[N,K]^T, double-buffered LDS fed by
// global_load_async_to_lds_b128. Block 128x128, 8 waves x (32x64).
// MODE 0: scatter f16 into q/k/v [4,16,2048,64]   MODE 1: f32 + bias
// =====================================================================
template <int MODE>
__global__ __launch_bounds__(256) void gemm_wmma_kernel(
    const f16* __restrict__ A, const f16* __restrict__ Bt, int K, int Ncols,
    f16* __restrict__ qb, f16* __restrict__ kb_, f16* __restrict__ vb,
    float* __restrict__ out, const float* __restrict__ bias) {
    __shared__ f16 lA[2][128 * 40];   // pitch 40 halves (80B rows)
    __shared__ f16 lB[2][128 * 40];

    const int t    = threadIdx.x;
    const int wave = t >> 5;
    const int lane = t & 31;
    const int l16  = lane & 15;
    const int hi   = lane >> 4;
    const int m0   = blockIdx.x * 128;
    const int n0   = blockIdx.y * 128;
    const int wm   = (wave & 3) * 32;
    const int wn   = (wave >> 2) * 64;

    // per-thread async-copy coordinates: rows r0 and r0+64, 8-half column chunk
    const int r0 = t >> 2;
    const int c4 = (t & 3) * 8;

    auto issue = [&](int buf, int kb0) {
        async_copy_b128(&lA[buf][r0 * 40 + c4],        A  + (size_t)(m0 + r0)      * K + kb0 + c4);
        async_copy_b128(&lA[buf][(r0 + 64) * 40 + c4], A  + (size_t)(m0 + r0 + 64) * K + kb0 + c4);
        async_copy_b128(&lB[buf][r0 * 40 + c4],        Bt + (size_t)(n0 + r0)      * K + kb0 + c4);
        async_copy_b128(&lB[buf][(r0 + 64) * 40 + c4], Bt + (size_t)(n0 + r0 + 64) * K + kb0 + c4);
    };

    v8f acc[2][4];
#pragma unroll
    for (int i = 0; i < 2; ++i)
#pragma unroll
        for (int j = 0; j < 4; ++j) acc[i][j] = {};

    issue(0, 0);
    for (int kb0 = 0; kb0 < K; kb0 += 32) {
        const int cur = (kb0 >> 5) & 1;
        wait_async0();          // own async writes for buf[cur] done
        __syncthreads();        // all waves' buf[cur] complete; prev reads retired
        if (kb0 + 32 < K) issue(cur ^ 1, kb0 + 32);   // overlap next slice

        Frag16 af[2], bf[4];
#pragma unroll
        for (int i = 0; i < 2; ++i)
            load_a_frag(lA[cur] + (wm + i * 16 + l16) * 40, hi, af[i]);
#pragma unroll
        for (int j = 0; j < 4; ++j)
            load_b_frag(lB[cur] + (wn + j * 16 + l16) * 40, hi, bf[j]);
#pragma unroll
        for (int i = 0; i < 2; ++i)
#pragma unroll
            for (int j = 0; j < 4; ++j)
                acc[i][j] = __builtin_amdgcn_wmma_f32_16x16x32_f16(
                    false, af[i].h, false, bf[j].h, (short)0, acc[i][j], false, false);
    }

    // epilogue (C/D layout: row = r + 8*hi within tile, col = lane%16)
#pragma unroll
    for (int i = 0; i < 2; ++i) {
#pragma unroll
        for (int j = 0; j < 4; ++j) {
#pragma unroll
            for (int r = 0; r < 8; ++r) {
                int m   = m0 + wm + i * 16 + r + hi * 8;
                int col = n0 + wn + j * 16 + l16;
                float val = acc[i][j][r];
                if (MODE == 0) {
                    int which = col >> 10;
                    int hc = col & 1023;
                    int h = hc >> 6, d = hc & 63;
                    int b = m >> 11, n = m & 2047;
                    size_t idx = (((size_t)(b * 16 + h)) * 2048 + n) * 64 + d;
                    f16* dst = (which == 0) ? qb : (which == 1) ? kb_ : vb;
                    dst[idx] = (f16)val;
                } else {
                    out[(size_t)m * Ncols + col] = val + bias[col];
                }
            }
        }
    }
}

// =====================================================================
// Flash attention: Q,K,V f16 [BH=64, N=2048, D=64] -> O f16 [B,N,H*D]
// grid (32 qtiles, 64 bh), 128 threads = 4 waves; wave owns a 16-row strip.
// =====================================================================
__global__ __launch_bounds__(128) void flash_attn_kernel(
    const f16* __restrict__ Q, const f16* __restrict__ Kk,
    const f16* __restrict__ V, f16* __restrict__ O) {
    const int t     = threadIdx.x;
    const int wave  = t >> 5;
    const int lane  = t & 31;
    const int l16   = lane & 15;
    const int hi    = lane >> 4;
    const int qtile = blockIdx.x;
    const int bh    = blockIdx.y;
    const size_t base = (size_t)bh * 2048 * 64;

    __shared__ f16 Vt[64 * 72];       // [d][j] (V^T tile), pitch 72 (144B rows)
    __shared__ f16 Pl[4 * 16 * 72];   // per-wave P strips [16][64], pitch 72

    // preload wave's Q strip as two A-fragments (d = 0..31, 32..63)
    Frag16 qa[2];
    {
        const f16* qrow = Q + base + (size_t)(qtile * 64 + wave * 16 + l16) * 64;
#pragma unroll
        for (int ks = 0; ks < 2; ++ks) load_a_frag(qrow + ks * 32, hi, qa[ks]);
    }

    v8f o[4];
#pragma unroll
    for (int dt = 0; dt < 4; ++dt) o[dt] = {};
    float mrow[8], lrow[8];
#pragma unroll
    for (int r = 0; r < 8; ++r) { mrow[r] = -1e30f; lrow[r] = 0.f; }

    for (int jt = 0; jt < 32; ++jt) {
        // ---- V tile -> LDS transposed ----
        {
            int nl = t >> 1;
            int dh = (t & 1) * 32;
            const uint4* vrow = (const uint4*)(V + base + (size_t)(jt * 64 + nl) * 64 + dh);
#pragma unroll
            for (int q4 = 0; q4 < 4; ++q4) {
                V4H8 w; w.v = vrow[q4];
#pragma unroll
                for (int e = 0; e < 8; ++e)
                    Vt[(dh + q4 * 8 + e) * 72 + nl] = w.h[e];
            }
        }
        // prefetch next K/V tile into cache hierarchy (global_prefetch_b8)
        if (jt + 1 < 32) {
            const size_t nxt = base + (size_t)((jt + 1) * 64 + (t & 63)) * 64;
            __builtin_prefetch(Kk + nxt, 0, 1);
            __builtin_prefetch(V + nxt, 0, 1);
        }
        __syncthreads();

        // ---- S = scale * Q K^T (B-fragments straight from global K rows) ----
        v8f s[4];
#pragma unroll
        for (int nt = 0; nt < 4; ++nt) {
            v8f c = {};
#pragma unroll
            for (int ks = 0; ks < 2; ++ks) {
                Frag16 bf;
                const f16* krow =
                    Kk + base + (size_t)(jt * 64 + nt * 16 + l16) * 64 + ks * 32;
                load_b_frag(krow, hi, bf);
                c = __builtin_amdgcn_wmma_f32_16x16x32_f16(
                    false, qa[ks].h, false, bf.h, (short)0, c, false, false);
            }
            s[nt] = c;
        }

        // ---- online softmax per row (8 rows/lane; 16 lanes share a row) ----
#pragma unroll
        for (int r = 0; r < 8; ++r) {
            float mx = -1e30f;
#pragma unroll
            for (int nt = 0; nt < 4; ++nt) {
                s[nt][r] *= 0.125f;
                mx = fmaxf(mx, s[nt][r]);
            }
#pragma unroll
            for (int off = 1; off < 16; off <<= 1)
                mx = fmaxf(mx, __shfl_xor(mx, off, 16));
            float mnew  = fmaxf(mrow[r], mx);
            float alpha = __expf(mrow[r] - mnew);
            mrow[r] = mnew;
            lrow[r] *= alpha;
#pragma unroll
            for (int dt = 0; dt < 4; ++dt) o[dt][r] *= alpha;
            float rsum = 0.f;
#pragma unroll
            for (int nt = 0; nt < 4; ++nt) {
                float p = __expf(s[nt][r] - mnew);
                s[nt][r] = p;
                rsum += p;
            }
#pragma unroll
            for (int off = 1; off < 16; off <<= 1)
                rsum += __shfl_xor(rsum, off, 16);
            lrow[r] += rsum;
            int prow = wave * 16 + r + hi * 8;
#pragma unroll
            for (int nt = 0; nt < 4; ++nt)
                Pl[prow * 72 + nt * 16 + l16] = (f16)s[nt][r];
        }
        __syncthreads();  // P + Vt ready

        // ---- O += P * V ----
#pragma unroll
        for (int dt = 0; dt < 4; ++dt) {
#pragma unroll
            for (int ks = 0; ks < 2; ++ks) {
                Frag16 pa, vf;
                load_a_frag(Pl + (wave * 16 + l16) * 72 + ks * 32, hi, pa);
                load_b_frag(Vt + (dt * 16 + l16) * 72 + ks * 32, hi, vf);
                o[dt] = __builtin_amdgcn_wmma_f32_16x16x32_f16(
                    false, pa.h, false, vf.h, (short)0, o[dt], false, false);
            }
        }
        __syncthreads();  // done reading Vt before next overwrite
    }

    // ---- epilogue: O /= l, write [b, n, h*64+d] f16 ----
    const int b = bh >> 4, h = bh & 15;
#pragma unroll
    for (int dt = 0; dt < 4; ++dt) {
#pragma unroll
        for (int r = 0; r < 8; ++r) {
            int n = qtile * 64 + wave * 16 + r + hi * 8;
            int d = dt * 16 + l16;
            float val = o[dt][r] / lrow[r];
            O[((size_t)(b * 2048 + n)) * 1024 + h * 64 + d] = (f16)val;
        }
    }
}

// =====================================================================
extern "C" void kernel_launch(void* const* d_in, const int* in_sizes, int n_in,
                              void* d_out, int out_size, void* d_ws, size_t ws_size,
                              hipStream_t stream) {
    const float* x     = (const float*)d_in[0];  // [4,2048,1024]
    const float* w_qkv = (const float*)d_in[1];  // [1024,3072]
    const float* w_out = (const float*)d_in[2];  // [1024,1024]
    const float* b_out = (const float*)d_in[3];  // [1024]
    float* out = (float*)d_out;                  // [4,2048,1024]

    size_t off = 0;
    auto alloc = [&](size_t bytes) -> void* {
        void* p = (char*)d_ws + off;
        off += (bytes + 255) & ~(size_t)255;
        return p;
    };
    f16* xh    = (f16*)alloc((size_t)8192 * 1024 * 2);
    f16* wqkvT = (f16*)alloc((size_t)3072 * 1024 * 2);
    f16* woutT = (f16*)alloc((size_t)1024 * 1024 * 2);
    f16* qb    = (f16*)alloc((size_t)64 * 2048 * 64 * 2);
    f16* kb    = (f16*)alloc((size_t)64 * 2048 * 64 * 2);
    f16* vb    = (f16*)alloc((size_t)64 * 2048 * 64 * 2);
    f16* ah    = (f16*)alloc((size_t)8192 * 1024 * 2);

    cvt_f16_kernel<<<(8192 * 1024) / 256, 256, 0, stream>>>(x, xh, 8192 * 1024);
    transpose_cvt_kernel<<<dim3(3072 / 32, 1024 / 32), 256, 0, stream>>>(w_qkv, wqkvT, 1024, 3072);
    transpose_cvt_kernel<<<dim3(1024 / 32, 1024 / 32), 256, 0, stream>>>(w_out, woutT, 1024, 1024);

    gemm_wmma_kernel<0><<<dim3(8192 / 128, 3072 / 128), 256, 0, stream>>>(
        xh, wqkvT, 1024, 3072, qb, kb, vb, nullptr, nullptr);

    flash_attn_kernel<<<dim3(2048 / 64, 64), 128, 0, stream>>>(qb, kb, vb, ah);

    gemm_wmma_kernel<1><<<dim3(8192 / 128, 1024 / 128), 256, 0, stream>>>(
        ah, woutT, 1024, 1024, nullptr, nullptr, nullptr, out, b_out);
}
